// ResolutionLevel_41369124995806
// MI455X (gfx1250) — matrix operand order repaired and checked by reading, not compiled, not run on hardware
//
#include <hip/hip_runtime.h>
#include <math.h>

// ---------------------------------------------------------------------------
// CDNA5 (gfx1250, wave32) fused implementation of the spatio-temporal
// cross-attention block.  All GEMMs go through v_wmma_f32_16x16x32_f16.
// ---------------------------------------------------------------------------

typedef __attribute__((ext_vector_type(16))) _Float16 v16h;
typedef __attribute__((ext_vector_type(8)))  float    v8f;
typedef _Float16 h16;

#define WMMA16(a, b, c) \
  __builtin_amdgcn_wmma_f32_16x16x32_f16(false, (a), false, (b), (short)0, (c), false, false)

#define N_Bt 1024
#define N_J  22

// ---- workspace byte offsets (prep-kernel outputs) --------------------------
#define WS_ASP   0         // f32 [22][22] normalized spatial adjacency
#define WS_STQ   2048      // f16 blob K=66->96,  N=512  (3*32*512 h)
#define WS_STK   100352    // f16 blob K=100->128,N=512  (4*32*512 h)
#define WS_STV   231424    // f16 blob K=100->128,N=512
#define WS_STO   362496    // f16 blob K=512,     N=66->80  (16*5*512 h)
#define WS_TSQ   444416    // f16 blob K=100->128,N=512
#define WS_TSK   575488    // f16 blob K=66->96,  N=512
#define WS_TSV   673792    // f16 blob K=66->96,  N=512
#define WS_TSO   772096    // f16 blob K=512,     N=100->112 (16*7*512 h)
#define WS_MTT   886784    // f16 blob (t_adj*mask)^T K=100->128, N=100->112
#define WS_FCT   915456    // f16 blob fc_w^T         K=100->128, N=100->112
#define WS_TOTAL 944128

// ---- dynamic LDS byte offsets ---------------------------------------------
#define SM_XS    0         // f32 [66][104]   x (residual)
#define SM_SP16  27456     // f16 [112][96]   spatial tokens [t][c]
#define SM_TP16  48960     // f16 [80][128]   temporal tokens [c][t]
#define SM_EHSP  69440     // f32 [112][80]   enh_sp accum [t][c]
#define SM_EHTP  105280    // f32 [80][112]   enh_tp accum [c][t]
#define SM_SCR   141120
#define SCR_QH   0         // f16 [112][64]
#define SCR_KH   14336     // f16 [112][64]
#define SCR_VHT  28672     // f16 [64][128]   V transposed [d][token]
#define SCR_SC   45056     // f32 [112][112]  scores
// aliases: probs f16[112][128] @ SCR_QH (28672B),  ctx16 f16[112][64] @ SCR_SC,
//          x16 / z16 f16[80][128] @ SCR_QH
#define SMEM_BYTES (141120 + 45056 + 50176)   // 236352 B  (< 320 KB/WGP)

// ---------------------------------------------------------------------------
// WMMA fragment loaders (wave32; A and B both use k = 16*(lane/16) + s, not the ISA table order)
// ---------------------------------------------------------------------------
__device__ __forceinline__ v16h a_frag(const h16* A, int lda, int row0, int k0) {
  // 16-bit A 16x32: lane L holds row L&15; halfword pair vi covers
  // K = (L>=16 ? 16 : 0) + vi*2 + {0,1}
  const int lane = threadIdx.x & 31;
  const h16* p = A + (row0 + (lane & 15)) * lda + k0 + ((lane >> 4) << 4);
  v16h a;
#pragma unroll
  for (int vi = 0; vi < 8; ++vi) {
    const int kb = vi << 1;
    a[2 * vi]     = p[kb];
    a[2 * vi + 1] = p[kb + 1];
  }
  return a;
}

// fragment-ready blob: frag (kcg,ntg) is 32 lanes x 16 halves, contiguous
__device__ __forceinline__ v16h b_blob(const h16* blob, int NT, int kcg, int ntg) {
  const v16h* p = (const v16h*)blob;
  return p[(kcg * NT + ntg) * 32 + (threadIdx.x & 31)];
}

// B[k][n] = M[n][k] where M is row-major [N][Kld]; per lane 16 contiguous halves
__device__ __forceinline__ v16h b_rowT(const h16* M, int ldm, int kc, int n0) {
  const int lane = threadIdx.x & 31;
  const v16h* p = (const v16h*)(M + (n0 + (lane & 15)) * ldm + kc * 32 + ((lane >> 4) << 4));
  return *p;
}

// ---------------------------------------------------------------------------
// GEMM building blocks (8 waves sweep 16x16 output tiles)
// ---------------------------------------------------------------------------
__device__ void gemm_blob(const h16* A, int lda, const h16* blob, int NT, int ntBase,
                          h16* D, int ldd, int mt, int ntl, int kch,
                          const float* bias, int nbBase, bool trans) {
  const int w = threadIdx.x >> 5, lane = threadIdx.x & 31;
  const int cc = lane & 15, g = lane >> 4;
  for (int tile = w; tile < mt * ntl; tile += 8) {
    const int mi = tile / ntl, ni = tile - mi * ntl;
    v8f acc = {};
    for (int kc = 0; kc < kch; ++kc) {
      v16h a = a_frag(A, lda, mi * 16, kc * 32);
      v16h b = b_blob(blob, NT, kc, ntBase + ni);
      acc = WMMA16(a, b, acc);
    }
    const int n = ni * 16 + cc;
    const float bv = bias ? bias[nbBase + n] : 0.f;
#pragma unroll
    for (int r = 0; r < 8; ++r) {
      const int m = mi * 16 + g * 8 + r;
      if (trans) D[n * ldd + m] = (h16)(acc[r] + bv);
      else       D[m * ldd + n] = (h16)(acc[r] + bv);
    }
  }
}

__device__ void gemm_scores(const h16* Q, const h16* K, float* S, int mt, int ntl) {
  const int w = threadIdx.x >> 5, lane = threadIdx.x & 31;
  const int cc = lane & 15, g = lane >> 4;
  for (int tile = w; tile < mt * ntl; tile += 8) {
    const int mi = tile / ntl, ni = tile - mi * ntl;
    v8f acc = {};
    for (int kc = 0; kc < 2; ++kc) {
      v16h a = a_frag(Q, 64, mi * 16, kc * 32);
      v16h b = b_rowT(K, 64, kc, ni * 16);     // B[k][n] = K[n][k]
      acc = WMMA16(a, b, acc);
    }
#pragma unroll
    for (int r = 0; r < 8; ++r)
      S[(mi * 16 + g * 8 + r) * 112 + ni * 16 + cc] = acc[r] * 0.125f;  // 1/sqrt(64)
  }
}

__device__ void softmax_rows(const float* S, h16* P, int Lq, int Lk, int Mpad, int Kpad) {
  const int row = threadIdx.x;
  if (row < Mpad) {
    h16* pr = P + row * 128;
    if (row < Lq) {
      const float* s = S + row * 112;
      float mx = -1e30f;
      for (int k = 0; k < Lk; ++k) mx = fmaxf(mx, s[k]);
      float sum = 0.f;
      for (int k = 0; k < Lk; ++k) sum += expf(s[k] - mx);
      const float inv = 1.f / sum;
      for (int k = 0; k < Kpad; ++k)
        pr[k] = (k < Lk) ? (h16)(expf(s[k] - mx) * inv) : (h16)0.f;
    } else {
      for (int k = 0; k < Kpad; ++k) pr[k] = (h16)0.f;
    }
  }
}

__device__ void gemm_ctx(const h16* P, const h16* VT, h16* ctx, int mt, int kch) {
  const int w = threadIdx.x >> 5, lane = threadIdx.x & 31;
  const int cc = lane & 15, g = lane >> 4;
  for (int tile = w; tile < mt * 4; tile += 8) {
    const int mi = tile >> 2, ni = tile & 3;
    v8f acc = {};
    for (int kc = 0; kc < kch; ++kc) {
      v16h a = a_frag(P, 128, mi * 16, kc * 32);
      v16h b = b_rowT(VT, 128, kc, ni * 16);   // B[k][d] = VT[d][k]
      acc = WMMA16(a, b, acc);
    }
#pragma unroll
    for (int r = 0; r < 8; ++r)
      ctx[(mi * 16 + g * 8 + r) * 64 + ni * 16 + cc] = (h16)acc[r];
  }
}

__device__ void gemm_outproj(const h16* ctx, const h16* blob, int NT, int kcBase,
                             float* Enh, int ldE, int mt, int ntl) {
  const int w = threadIdx.x >> 5, lane = threadIdx.x & 31;
  const int cc = lane & 15, g = lane >> 4;
  for (int tile = w; tile < mt * ntl; tile += 8) {
    const int mi = tile / ntl, ni = tile - mi * ntl;
    v8f acc;
#pragma unroll
    for (int r = 0; r < 8; ++r)
      acc[r] = Enh[(mi * 16 + g * 8 + r) * ldE + ni * 16 + cc];
    for (int kc = 0; kc < 2; ++kc) {
      v16h a = a_frag(ctx, 64, mi * 16, kc * 32);
      v16h b = b_blob(blob, NT, kcBase + kc, ni);
      acc = WMMA16(a, b, acc);
    }
#pragma unroll
    for (int r = 0; r < 8; ++r)
      Enh[(mi * 16 + g * 8 + r) * ldE + ni * 16 + cc] = acc[r];
  }
}

// ---------------------------------------------------------------------------
// Prep kernel: normalized adjacency + zero-padded fragment-ready weight blobs
// ---------------------------------------------------------------------------
__device__ void build_blob(h16* dst, const float* W, const float* W2,
                           int Ksrc, int Nsrc, int kch, int NT, int mode,
                           int tid, int stride) {
  const int total = kch * NT * 512;
  for (int e = tid; e < total; e += stride) {
    const int j    = e & 15;
    const int lane = (e >> 4) & 31;
    const int tile = e >> 9;
    const int kc = tile / NT, ntg = tile - kc * NT;
    const int k = kc * 32 + ((lane >> 4) << 4) + j;
    const int n = ntg * 16 + (lane & 15);
    float v = 0.f;
    if (k < Ksrc && n < Nsrc) {
      if (mode == 0)      v = W[k * Nsrc + n];                       // B[k][n] = W[k][n]
      else if (mode == 1) v = W[n * Ksrc + k];                       // B[k][n] = W[n][k]
      else                v = W[n * Ksrc + k] * W2[n * Ksrc + k];    // transposed product
    }
    dst[e] = (h16)v;
  }
}

__global__ void prep_kernel(const float* __restrict__ adj, const float* __restrict__ sadj,
                            const float* __restrict__ trajm, const float* __restrict__ tadj,
                            const float* __restrict__ stwq, const float* __restrict__ stwk,
                            const float* __restrict__ stwv, const float* __restrict__ stwo,
                            const float* __restrict__ tswq, const float* __restrict__ tswk,
                            const float* __restrict__ tswv, const float* __restrict__ tswo,
                            const float* __restrict__ fcw, char* __restrict__ ws) {
  const int tid = blockIdx.x * blockDim.x + threadIdx.x;
  const int stride = gridDim.x * blockDim.x;
  float* Asp = (float*)(ws + WS_ASP);
  for (int i = tid; i < N_J * N_J; i += stride) {
    const int v = i / N_J, j = i - v * N_J;
    float dv = 0.f, dj = 0.f;
    for (int q = 0; q < N_J; ++q) { dv += adj[v * N_J + q]; dj += adj[j * N_J + q]; }
    const float iv = dv > 0.f ? rsqrtf(dv) : 0.f;
    const float ij = dj > 0.f ? rsqrtf(dj) : 0.f;
    Asp[i] = sadj[i] * adj[i] * iv * ij;
  }
  build_blob((h16*)(ws + WS_STQ), stwq, nullptr, 66, 512, 3, 32, 0, tid, stride);
  build_blob((h16*)(ws + WS_STK), stwk, nullptr, 100, 512, 4, 32, 0, tid, stride);
  build_blob((h16*)(ws + WS_STV), stwv, nullptr, 100, 512, 4, 32, 0, tid, stride);
  build_blob((h16*)(ws + WS_STO), stwo, nullptr, 512, 66, 16, 5, 0, tid, stride);
  build_blob((h16*)(ws + WS_TSQ), tswq, nullptr, 100, 512, 4, 32, 0, tid, stride);
  build_blob((h16*)(ws + WS_TSK), tswk, nullptr, 66, 512, 3, 32, 0, tid, stride);
  build_blob((h16*)(ws + WS_TSV), tswv, nullptr, 66, 512, 3, 32, 0, tid, stride);
  build_blob((h16*)(ws + WS_TSO), tswo, nullptr, 512, 100, 16, 7, 0, tid, stride);
  build_blob((h16*)(ws + WS_MTT), tadj, trajm, 100, 100, 4, 7, 2, tid, stride); // (t_adj*mask)^T
  build_blob((h16*)(ws + WS_FCT), fcw, nullptr, 100, 100, 4, 7, 1, tid, stride); // fc_w^T
}

// ---------------------------------------------------------------------------
// Fused per-batch kernel: 1 block (256 thr, 8 wave32) per batch element
// ---------------------------------------------------------------------------
__global__ __launch_bounds__(256, 1) void
fused_kernel(const float* __restrict__ xg, const char* __restrict__ ws,
             const float* __restrict__ stbq, const float* __restrict__ stbk,
             const float* __restrict__ stbv, const float* __restrict__ stbo,
             const float* __restrict__ tsbq, const float* __restrict__ tsbk,
             const float* __restrict__ tsbv, const float* __restrict__ tsbo,
             const float* __restrict__ lnA, const float* __restrict__ lnB,
             const float* __restrict__ fcb, float* __restrict__ outg) {
  extern __shared__ __align__(128) char smem[];
  __shared__ float sAsp[N_J * N_J];

  const int tid = threadIdx.x;
  const int b = blockIdx.x;

  float* xs   = (float*)(smem + SM_XS);
  h16*  sp16  = (h16*)(smem + SM_SP16);
  h16*  tp16  = (h16*)(smem + SM_TP16);
  float* ehsp = (float*)(smem + SM_EHSP);
  float* ehtp = (float*)(smem + SM_EHTP);
  char*  scr  = smem + SM_SCR;
  h16*  Qh    = (h16*)(scr + SCR_QH);
  h16*  Kh    = (h16*)(scr + SCR_KH);
  h16*  VhT   = (h16*)(scr + SCR_VHT);
  float* scF  = (float*)(scr + SCR_SC);
  h16*  probs = (h16*)(scr + SCR_QH);   // alias Qh+Kh (dead when written)
  h16*  ctx16 = (h16*)(scr + SCR_SC);   // alias scores (dead when written)
  h16*  x16   = (h16*)(scr + SCR_QH);   // phase-0 alias
  h16*  z16   = (h16*)(scr + SCR_QH);   // final-phase alias

  const float* AspG = (const float*)(ws + WS_ASP);
  const h16* WstQ = (const h16*)(ws + WS_STQ);
  const h16* WstK = (const h16*)(ws + WS_STK);
  const h16* WstV = (const h16*)(ws + WS_STV);
  const h16* WstO = (const h16*)(ws + WS_STO);
  const h16* WtsQ = (const h16*)(ws + WS_TSQ);
  const h16* WtsK = (const h16*)(ws + WS_TSK);
  const h16* WtsV = (const h16*)(ws + WS_TSV);
  const h16* WtsO = (const h16*)(ws + WS_TSO);
  const h16* MtT  = (const h16*)(ws + WS_MTT);
  const h16* FcT  = (const h16*)(ws + WS_FCT);

  // ---- phase 0: load x, zero pads ----------------------------------------
  for (int i = tid; i < 112 * 96; i += 256) sp16[i] = (h16)0.f;
  for (int i = tid; i < 80 * 16; i += 256) {               // tp16 cols 112..127
    const int r = i >> 4;
    tp16[r * 128 + 112 + (i & 15)] = (h16)0.f;
  }
  for (int i = tid; i < 80 * 128; i += 256) {              // x16 zero-padded
    const int cc = i >> 7, t = i & 127;
    float v = 0.f;
    if (cc < 66 && t < 100) { v = xg[(size_t)b * 6600 + cc * 100 + t]; xs[cc * 104 + t] = v; }
    x16[i] = (h16)v;
  }
  for (int i = tid; i < N_J * N_J; i += 256) sAsp[i] = AspG[i];
  __syncthreads();

  // temporal = x @ (t_adj*mask)^T   (WMMA)  -> tp16 [c][t]
  gemm_blob(x16, 128, MtT, 7, 0, tp16, 128, 5, 7, 4, nullptr, 0, false);
  // spatial tokens sp16[t][c] = sum_j A[v][j] * x[j*3+d][t]  (tiny, VALU)
  for (int i = tid; i < 6600; i += 256) {
    const int cc = i / 100, t = i - cc * 100;
    const int v = cc / 3, d = cc - v * 3;
    float s = 0.f;
    for (int j = 0; j < N_J; ++j) s += sAsp[v * N_J + j] * xs[(j * 3 + d) * 104 + t];
    sp16[t * 96 + cc] = (h16)s;
  }
  // init enhancement accumulators with output-proj biases
  for (int i = tid; i < 112 * 80; i += 256) { const int cc = i % 80; ehsp[i] = (cc < 66) ? stbo[cc] : 0.f; }
  for (int i = tid; i < 80 * 112; i += 256) { const int t = i % 112; ehtp[i] = (t < 100) ? tsbo[t] : 0.f; }
  __syncthreads();

  // ---- spatial->temporal attention (Lq=100, Lk=66), head by head ---------
  for (int h = 0; h < 8; ++h) {
    for (int i = tid; i < 64 * 128; i += 256) VhT[i] = (h16)0.f;
    __syncthreads();
    gemm_blob(sp16, 96,  WstQ, 32, h * 4, Qh,  64,  7, 4, 3, stbq, h * 64, false);
    gemm_blob(tp16, 128, WstK, 32, h * 4, Kh,  64,  5, 4, 4, stbk, h * 64, false);
    gemm_blob(tp16, 128, WstV, 32, h * 4, VhT, 128, 5, 4, 4, stbv, h * 64, true);
    __syncthreads();
    gemm_scores(Qh, Kh, scF, 7, 5);
    __syncthreads();
    softmax_rows(scF, probs, 100, 66, 112, 96);
    __syncthreads();
    gemm_ctx(probs, VhT, ctx16, 7, 3);
    __syncthreads();
    gemm_outproj(ctx16, WstO, 5, h * 2, ehsp, 80, 7, 5);
    __syncthreads();
  }

  // ---- temporal->spatial attention (Lq=66, Lk=100) -----------------------
  for (int h = 0; h < 8; ++h) {
    for (int i = tid; i < 64 * 128; i += 256) VhT[i] = (h16)0.f;
    __syncthreads();
    gemm_blob(tp16, 128, WtsQ, 32, h * 4, Qh,  64,  5, 4, 4, tsbq, h * 64, false);
    gemm_blob(sp16, 96,  WtsK, 32, h * 4, Kh,  64,  7, 4, 3, tsbk, h * 64, false);
    gemm_blob(sp16, 96,  WtsV, 32, h * 4, VhT, 128, 7, 4, 3, tsbv, h * 64, true);
    __syncthreads();
    gemm_scores(Qh, Kh, scF, 5, 7);
    __syncthreads();
    softmax_rows(scF, probs, 66, 100, 80, 128);
    __syncthreads();
    gemm_ctx(probs, VhT, ctx16, 5, 4);
    __syncthreads();
    gemm_outproj(ctx16, WtsO, 7, h * 2, ehtp, 112, 5, 7);
    __syncthreads();
  }

  // ---- combine + LayerNorm(channel) + residual -> z16 --------------------
  for (int i = tid; i < 80 * 128; i += 256) z16[i] = (h16)0.f;
  __syncthreads();
  for (int t = tid; t < 100; t += 256) {
    float sum = 0.f, sq = 0.f;
    for (int cc = 0; cc < 66; ++cc) {
      const float v = ehsp[t * 80 + cc] + ehtp[cc * 112 + t];
      sum += v; sq += v * v;
    }
    const float mean = sum * (1.f / 66.f);
    const float var  = sq * (1.f / 66.f) - mean * mean;
    const float inv  = rsqrtf(var + 1e-5f);
    for (int cc = 0; cc < 66; ++cc) {
      const float v = ehsp[t * 80 + cc] + ehtp[cc * 112 + t];
      const float y = (v - mean) * inv * lnA[cc] + lnB[cc];
      z16[cc * 128 + t] = (h16)(y + xs[cc * 104 + t]);
    }
  }
  __syncthreads();

  // ---- out = tanh(z @ fc_w^T + fc_b)  (WMMA) -----------------------------
  {
    const int w = tid >> 5, lane = tid & 31;
    const int cc = lane & 15, g = lane >> 4;
    for (int tile = w; tile < 35; tile += 8) {
      const int mi = tile / 7, ni = tile - mi * 7;
      v8f acc = {};
      for (int kc = 0; kc < 4; ++kc) {
        v16h a = a_frag(z16, 128, mi * 16, kc * 32);
        v16h bb = b_blob(FcT, 7, kc, ni);
        acc = WMMA16(a, bb, acc);
      }
      const int n = ni * 16 + cc;
      if (n < 100) {
        const float bv = fcb[n];
#pragma unroll
        for (int r = 0; r < 8; ++r) {
          const int m = mi * 16 + g * 8 + r;
          if (m < 66) outg[(size_t)b * 6600 + m * 100 + n] = tanhf(acc[r] + bv);
        }
      }
    }
  }
}

// ---------------------------------------------------------------------------
extern "C" void kernel_launch(void* const* d_in, const int* in_sizes, int n_in,
                              void* d_out, int out_size, void* d_ws, size_t ws_size,
                              hipStream_t stream) {
  const float* x     = (const float*)d_in[0];
  const float* adj   = (const float*)d_in[1];
  const float* sadj  = (const float*)d_in[2];
  const float* trajm = (const float*)d_in[3];
  const float* tadj  = (const float*)d_in[4];
  const float* stwq  = (const float*)d_in[5];
  const float* stbq  = (const float*)d_in[6];
  const float* stwk  = (const float*)d_in[7];
  const float* stbk  = (const float*)d_in[8];
  const float* stwv  = (const float*)d_in[9];
  const float* stbv  = (const float*)d_in[10];
  const float* stwo  = (const float*)d_in[11];
  const float* stbo  = (const float*)d_in[12];
  const float* tswq  = (const float*)d_in[13];
  const float* tsbq  = (const float*)d_in[14];
  const float* tswk  = (const float*)d_in[15];
  const float* tsbk  = (const float*)d_in[16];
  const float* tswv  = (const float*)d_in[17];
  const float* tsbv  = (const float*)d_in[18];
  const float* tswo  = (const float*)d_in[19];
  const float* tsbo  = (const float*)d_in[20];
  const float* lnA   = (const float*)d_in[21];
  const float* lnB   = (const float*)d_in[22];
  const float* fcw   = (const float*)d_in[23];
  const float* fcb   = (const float*)d_in[24];
  char* ws = (char*)d_ws;

  (void)hipFuncSetAttribute((const void*)fused_kernel,
                            hipFuncAttributeMaxDynamicSharedMemorySize, SMEM_BYTES);

  prep_kernel<<<96, 256, 0, stream>>>(adj, sadj, trajm, tadj,
                                      stwq, stwk, stwv, stwo,
                                      tswq, tswk, tswv, tswo, fcw, ws);
  fused_kernel<<<N_Bt, 256, SMEM_BYTES, stream>>>(x, ws,
                                                  stbq, stbk, stbv, stbo,
                                                  tsbq, tsbk, tsbv, tsbo,
                                                  lnA, lnB, fcb, (float*)d_out);
}
